// LoRALinear_89644557402583
// MI455X (gfx1250) — matrix operand order repaired and checked by reading, not compiled, not run on hardware
//
#include <hip/hip_runtime.h>
#include <cstdint>
#include <cstddef>

// ---- CDNA5 vector types (plain ext_vector, trivially union-able) ----
typedef __attribute__((ext_vector_type(16))) __bf16        v16bf;
typedef __attribute__((ext_vector_type(2)))  __bf16        v2bf;
typedef __attribute__((ext_vector_type(8)))  float         v8f;
typedef __attribute__((ext_vector_type(4)))  float         f32x4;
typedef __attribute__((ext_vector_type(4)))  unsigned int  u32x4;

// ---- problem constants (fixed by the reference) ----
#define M_TOK 8192          // B*S = 4*2048
#define N_OUT 4096
#define K_IN  4096
#define R_PAD 32            // LoRA rank 16 padded to one bf16 WMMA K-step
#define LSCALE 2.0f         // ALPHA/RANK = 32/16

constexpr int BM = 128, BN = 256, BK = 32;
constexpr int NKT = K_IN / BK;          // 128 dense k-steps; index NKT = LoRA tail step
constexpr int APITCH = 40;              // ushorts per LDS row: 32 data + 8 pad (80B, 16B aligned)

// float pair -> packed bf16 dword using the native conversion hardware
// (clang lowers the fptrunc pair to packed v_cvt on gfx1250)
__device__ __forceinline__ unsigned pk2(float lo, float hi) {
  union { v2bf v; unsigned u; } c;
  c.v.x = (__bf16)lo;
  c.v.y = (__bf16)hi;
  return c.u;
}
__device__ __forceinline__ unsigned f2bf_lo(float f) {
  union { v2bf v; unsigned u; } c;
  c.v.x = (__bf16)f;
  c.v.y = (__bf16)0.f;
  return c.u & 0xFFFFu;
}

// =====================================================================
// Pass 1: xa[m][r] = LSCALE * sum_k x[m,k] * lora_A[r,k]  (bf16, r padded to 32)
// One wave32 per token row; butterfly reduce across the wave.
// =====================================================================
__global__ __launch_bounds__(256) void xa_kernel(const float* __restrict__ x,
                                                 const float* __restrict__ lA,
                                                 unsigned short* __restrict__ xa) {
  const int lane = threadIdx.x & 31;
  const int wave = threadIdx.x >> 5;
  const int m = blockIdx.x * 8 + wave;
  float acc[16];
#pragma unroll
  for (int r = 0; r < 16; ++r) acc[r] = 0.f;
  const float* xr = x + (size_t)m * K_IN;
  for (int k = lane; k < K_IN; k += 32) {
    float xv = xr[k];
#pragma unroll
    for (int r = 0; r < 16; ++r) acc[r] = fmaf(xv, lA[r * K_IN + k], acc[r]);
  }
  float mine = 0.f;
#pragma unroll
  for (int r = 0; r < 16; ++r) {
    float v = acc[r];
#pragma unroll
    for (int off = 16; off > 0; off >>= 1) v += __shfl_xor(v, off);
    if (lane == r) mine = v;
  }
  unsigned short o = (lane < 16) ? (unsigned short)f2bf_lo(mine * LSCALE)
                                 : (unsigned short)0;
  xa[(size_t)m * R_PAD + lane] = o;
}

// =====================================================================
// Pass 2: out = x @ W^T + b + xa @ lora_B^T   (bf16 WMMA, fp32 accum)
// =====================================================================
struct TileRegs { unsigned a[8]; unsigned b[2][8]; };

__device__ __forceinline__ void load_tile(int ks, int m0, int n0, int tid,
    const float* __restrict__ x, const float* __restrict__ W,
    const float* __restrict__ lB, const unsigned short* __restrict__ xa,
    TileRegs& t) {
  const int ar = tid >> 1, ah = tid & 1;
  if (ks < NKT) {
    const int k0 = ks * BK;
    // A tile: 128 rows of x, 32 k's; this thread: row ar, half ah (16 floats)
    const f32x4* p = (const f32x4*)(x + (size_t)(m0 + ar) * K_IN + k0 + ah * 16);
    f32x4 v0 = p[0], v1 = p[1], v2 = p[2], v3 = p[3];
    t.a[0]=pk2(v0.x,v0.y); t.a[1]=pk2(v0.z,v0.w);
    t.a[2]=pk2(v1.x,v1.y); t.a[3]=pk2(v1.z,v1.w);
    t.a[4]=pk2(v2.x,v2.y); t.a[5]=pk2(v2.z,v2.w);
    t.a[6]=pk2(v3.x,v3.y); t.a[7]=pk2(v3.z,v3.w);
    // B tile: 256 rows of W (B[k][n] = W[n][k] -> contiguous rows), 2 rows/thread
#pragma unroll
    for (int i = 0; i < 2; ++i) {
      const int idx = tid + 256 * i;
      const int br = idx >> 1, bh = idx & 1;
      const f32x4* q = (const f32x4*)(W + (size_t)(n0 + br) * K_IN + k0 + bh * 16);
      f32x4 w0 = q[0], w1 = q[1], w2 = q[2], w3 = q[3];
      t.b[i][0]=pk2(w0.x,w0.y); t.b[i][1]=pk2(w0.z,w0.w);
      t.b[i][2]=pk2(w1.x,w1.y); t.b[i][3]=pk2(w1.z,w1.w);
      t.b[i][4]=pk2(w2.x,w2.y); t.b[i][5]=pk2(w2.z,w2.w);
      t.b[i][6]=pk2(w3.x,w3.y); t.b[i][7]=pk2(w3.z,w3.w);
    }
    // prefetch k+2 tile into caches (global_prefetch_b8)
    if (ks + 2 < NKT) {
      __builtin_prefetch(x + (size_t)(m0 + ar) * K_IN + (ks + 2) * BK + ah * 16, 0, 1);
      __builtin_prefetch(W + (size_t)(n0 + ar) * K_IN + (ks + 2) * BK + ah * 16, 0, 1);
    }
  } else {
    // LoRA tail k-step: A side = pre-scaled bf16 xa (already padded to K=32)
    const u32x4* q = (const u32x4*)(xa + (size_t)(m0 + ar) * R_PAD + ah * 16);
    u32x4 u0 = q[0], u1 = q[1];
    t.a[0]=u0.x; t.a[1]=u0.y; t.a[2]=u0.z; t.a[3]=u0.w;
    t.a[4]=u1.x; t.a[5]=u1.y; t.a[6]=u1.z; t.a[7]=u1.w;
    // B side = lora_B rows (16 floats), zero-padded to 32
#pragma unroll
    for (int i = 0; i < 2; ++i) {
      const int idx = tid + 256 * i;
      const int br = idx >> 1, bh = idx & 1;
      if (bh == 0) {
        const f32x4* q2 = (const f32x4*)(lB + (size_t)(n0 + br) * 16);
        f32x4 w0 = q2[0], w1 = q2[1], w2 = q2[2], w3 = q2[3];
        t.b[i][0]=pk2(w0.x,w0.y); t.b[i][1]=pk2(w0.z,w0.w);
        t.b[i][2]=pk2(w1.x,w1.y); t.b[i][3]=pk2(w1.z,w1.w);
        t.b[i][4]=pk2(w2.x,w2.y); t.b[i][5]=pk2(w2.z,w2.w);
        t.b[i][6]=pk2(w3.x,w3.y); t.b[i][7]=pk2(w3.z,w3.w);
      } else {
#pragma unroll
        for (int j = 0; j < 8; ++j) t.b[i][j] = 0u;
      }
    }
  }
}

__device__ __forceinline__ void store_tile(int buf, int tid,
    unsigned short* sA, unsigned short* sB, const TileRegs& t) {
  const int ar = tid >> 1, ah = tid & 1;
  {
    u32x4* pa = (u32x4*)(sA + (size_t)buf * (BM * APITCH) + ar * APITCH + ah * 16);
    u32x4 lo = { t.a[0], t.a[1], t.a[2], t.a[3] };
    u32x4 hi = { t.a[4], t.a[5], t.a[6], t.a[7] };
    pa[0] = lo; pa[1] = hi;
  }
#pragma unroll
  for (int i = 0; i < 2; ++i) {
    const int idx = tid + 256 * i;
    const int br = idx >> 1, bh = idx & 1;
    u32x4* pb = (u32x4*)(sB + (size_t)buf * (BN * APITCH) + br * APITCH + bh * 16);
    u32x4 lo = { t.b[i][0], t.b[i][1], t.b[i][2], t.b[i][3] };
    u32x4 hi = { t.b[i][4], t.b[i][5], t.b[i][6], t.b[i][7] };
    pb[0] = lo; pb[1] = hi;
  }
}

union Frag { u32x4 u[2]; v16bf v; };

__global__ __launch_bounds__(256) void lora_gemm(const float* __restrict__ x,
    const float* __restrict__ W, const float* __restrict__ bias,
    const float* __restrict__ lB, const unsigned short* __restrict__ xa,
    float* __restrict__ out) {
  __shared__ __align__(16) unsigned short sA[2 * BM * APITCH];   // 20 KB
  __shared__ __align__(16) unsigned short sB[2 * BN * APITCH];   // 40 KB

  const int tid  = threadIdx.x;
  const int m0   = blockIdx.y * BM;
  const int n0   = blockIdx.x * BN;
  const int wave = tid >> 5, lane = tid & 31;
  const int wm = wave >> 2, wn = wave & 3;       // 2 x 4 waves -> 64x64 per wave
  const int lr = lane & 15, lh = lane >> 4;

  v8f c[4][4] = {};
  TileRegs t;
  load_tile(0, m0, n0, tid, x, W, lB, xa, t);
  store_tile(0, tid, sA, sB, t);

  for (int ks = 0; ks <= NKT; ++ks) {
    __syncthreads();
    if (ks < NKT) load_tile(ks + 1, m0, n0, tid, x, W, lB, xa, t);

    const int buf = ks & 1;
    // A fragments: lane lr holds row M=lr; K chunks {lh*8..+7, lh*8+16..+23} (ISA 7.12.2)
    Frag fa[4], fb[4];
    const unsigned short* baseA = sA + (size_t)buf * (BM * APITCH) + (wm * 64) * APITCH;
#pragma unroll
    for (int i = 0; i < 4; ++i) {
      const u32x4* p = (const u32x4*)(baseA + (i * 16 + lr) * APITCH + lh * 8);
      fa[i].u[0] = p[0]; fa[i].u[1] = p[2];      // chunk2 at +16 elems (+32 B)
    }
    // B fragments: lane lr holds column N=lr (= W row, contiguous); K chunks {lh*8..+7, lh*8+16..+23}
    const unsigned short* baseB = sB + (size_t)buf * (BN * APITCH) + (wn * 64) * APITCH;
#pragma unroll
    for (int j = 0; j < 4; ++j) {
      const u32x4* p = (const u32x4*)(baseB + (j * 16 + lr) * APITCH + lh * 8);
      fb[j].u[0] = p[0]; fb[j].u[1] = p[2];
    }
#pragma unroll
    for (int i = 0; i < 4; ++i)
#pragma unroll
      for (int j = 0; j < 4; ++j)
        c[i][j] = __builtin_amdgcn_wmma_f32_16x16x32_bf16(
            false, fa[i].v, false, fb[j].v, (short)0, c[i][j], false, false);

    if (ks < NKT) {
      __syncthreads();
      store_tile((ks + 1) & 1, tid, sA, sB, t);
    }
  }

  // Epilogue: C layout -> lane holds column n = lr; VGPR v holds row v + 8*lh. Add bias.
#pragma unroll
  for (int j = 0; j < 4; ++j) {
    const int n = n0 + wn * 64 + j * 16 + lr;
    const float bv = bias[n];
#pragma unroll
    for (int i = 0; i < 4; ++i) {
      const int mb = m0 + wm * 64 + i * 16 + lh * 8;
#pragma unroll
      for (int v = 0; v < 8; ++v)
        out[(size_t)(mb + v) * N_OUT + n] = c[i][j][v] + bv;
    }
  }
}

// =====================================================================
extern "C" void kernel_launch(void* const* d_in, const int* in_sizes, int n_in,
                              void* d_out, int out_size, void* d_ws, size_t ws_size,
                              hipStream_t stream) {
  (void)in_sizes; (void)n_in; (void)out_size; (void)ws_size;
  const float* x  = (const float*)d_in[0];        // [8192, 4096]
  const float* W  = (const float*)d_in[1];        // [4096, 4096]
  const float* b  = (const float*)d_in[2];        // [4096]
  const float* lA = (const float*)d_in[3];        // [16, 4096]
  const float* lB = (const float*)d_in[4];        // [4096, 16]
  float* out = (float*)d_out;                     // [8192, 4096]
  unsigned short* xa = (unsigned short*)d_ws;     // 8192*32 bf16 = 512 KB scratch

  xa_kernel<<<dim3(M_TOK / 8), dim3(256), 0, stream>>>(x, lA, xa);
  lora_gemm<<<dim3(N_OUT / BN, M_TOK / BM), dim3(256), 0, stream>>>(x, W, b, lB, xa, out);
}
